// Qint_Linear_21457656611364
// MI455X (gfx1250) — matrix operand order repaired and checked by reading, not compiled, not run on hardware
//
#include <hip/hip_runtime.h>
#include <stdint.h>

typedef __attribute__((ext_vector_type(16))) _Float16 v16h;
typedef __attribute__((ext_vector_type(8)))  _Float16 v8h;
typedef __attribute__((ext_vector_type(8)))  float    v8f;
typedef __attribute__((ext_vector_type(4)))  float    v4f;
typedef __attribute__((ext_vector_type(4)))  unsigned int v4u;
typedef __attribute__((ext_vector_type(8)))  int      v8i;
typedef __attribute__((ext_vector_type(4)))  int      v4i;

#define TOKENS 8192
#define IN_F   4096
#define OUT_F  11008

#define BM  128   // tokens per block
#define BN  128   // out-features per block
#define BK  32    // K per stage (fallback kernel)
#define BK2 64    // K per stage (TDM kernel: 2 WMMA K-steps per barrier)

union HV { v16h v; v8h h[2]; };

// ---------------------------------------------------------------------------
// TDM: issue a 2-D tile DMA  global(f16) -> LDS  via the Tensor Data Mover.
// D# layout per CDNA5 ISA 8.3-8.5. Tile rows land contiguously in LDS
// (row-major, tile_d0 elements per row).
// ---------------------------------------------------------------------------
__device__ __forceinline__ void tdm_load_2d_f16(
    uint32_t lds_off_bytes, const _Float16* gptr,
    uint32_t tensor_d0, uint32_t tensor_d1,
    uint32_t tile_d0, uint32_t tile_d1,
    uint32_t stride0_elems)
{
    uint64_t ga = (uint64_t)(uintptr_t)gptr;
    v4u g0;
    g0.x = 1u;                                   // count=1, user descriptor
    g0.y = lds_off_bytes;                        // lds_addr
    g0.z = (uint32_t)ga;                         // global_addr[31:0]
    g0.w = (uint32_t)(ga >> 32) | (2u << 30);    // global_addr[56:32] | type=2

    v8i g1;
    g1[0] = (int)(1u << 16);                     // data_size=1 (2 bytes/elem)
    g1[1] = (int)((tensor_d0 & 0xFFFFu) << 16);  // tensor_dim0[15:0]
    g1[2] = (int)((tensor_d0 >> 16) | ((tensor_d1 & 0xFFFFu) << 16));
    g1[3] = (int)((tensor_d1 >> 16) | ((tile_d0 & 0xFFFFu) << 16));
    g1[4] = (int)(tile_d1 & 0xFFFFu);            // tile_dim1; tile_dim2=0
    g1[5] = (int)stride0_elems;                  // tensor_dim0_stride[31:0]
    g1[6] = 0;
    g1[7] = 0;

    v4i z4 = {0, 0, 0, 0};
#if defined(__clang_major__) && (__clang_major__ >= 23)
    v8i z8 = {0, 0, 0, 0, 0, 0, 0, 0};
    __builtin_amdgcn_tensor_load_to_lds(g0, g1, z4, z4, z8, 0);
#else
    __builtin_amdgcn_tensor_load_to_lds(g0, g1, z4, z4, 0);
#endif
}

// ---------------------------------------------------------------------------
// One-time converts into workspace (f16 copies; int8 -> f16 is exact)
// ---------------------------------------------------------------------------
__global__ __launch_bounds__(256) void cvt_x_f16(const float* __restrict__ x,
                                                 _Float16* __restrict__ xh)
{
    size_t i = ((size_t)blockIdx.x * 256 + threadIdx.x) * 8;
    v4f f0 = *(const v4f*)(x + i);
    v4f f1 = *(const v4f*)(x + i + 4);
    v8h h;
    #pragma unroll
    for (int j = 0; j < 4; ++j) {
        h[j]     = (_Float16)f0[j];
        h[j + 4] = (_Float16)f1[j];
    }
    *(v8h*)(xh + i) = h;
}

__global__ __launch_bounds__(256) void cvt_w_f16(const int8_t* __restrict__ w,
                                                 _Float16* __restrict__ wh)
{
    size_t i = ((size_t)blockIdx.x * 256 + threadIdx.x) * 16;
    int4 wv = *(const int4*)(w + i);
    const int8_t* wb = (const int8_t*)&wv;
    v8h h0, h1;
    #pragma unroll
    for (int j = 0; j < 8; ++j) {
        h0[j] = (_Float16)(int)wb[j];
        h1[j] = (_Float16)(int)wb[j + 8];
    }
    *(v8h*)(wh + i)     = h0;
    *(v8h*)(wh + i + 8) = h1;
}

// ---------------------------------------------------------------------------
// Main GEMM: TDM-staged, double-buffered f16 WMMA.
// Stage s+1's tile DMAs are in flight (TENSORcnt=2) while stage s runs its
// 16 WMMAs; s_wait_tensorcnt(2) proves stage s's tiles landed.
// ---------------------------------------------------------------------------
__global__ __launch_bounds__(256) void qlinear_wmma_tdm(
    const _Float16* __restrict__ xh,   // [TOKENS][IN_F]
    const _Float16* __restrict__ wh,   // [OUT_F][IN_F]
    const float* __restrict__ scales,
    const float* __restrict__ bias,
    float* __restrict__ out)
{
    __shared__ _Float16 lds_a[2][BM * BK2];   // 2 x 16 KB, [m][k]
    __shared__ _Float16 lds_b[2][BN * BK2];   // 2 x 16 KB, [n][k]

    const int tid   = threadIdx.x;
    const int lane  = tid & 31;
    const int wave  = tid >> 5;        // 0..7
    const int waveM = wave >> 1;       // 0..3 -> 32 rows
    const int waveN = wave & 1;        // 0..1 -> 64 cols

    const int m0 = blockIdx.y * BM;
    const int n0 = blockIdx.x * BN;

    const uint32_t a_off[2] = { (uint32_t)(uintptr_t)&lds_a[0][0],
                                (uint32_t)(uintptr_t)&lds_a[1][0] };
    const uint32_t b_off[2] = { (uint32_t)(uintptr_t)&lds_b[0][0],
                                (uint32_t)(uintptr_t)&lds_b[1][0] };

    const _Float16* xrow = xh + (size_t)m0 * IN_F;
    const _Float16* wrow = wh + (size_t)n0 * IN_F;

    v8f acc[2][4] = {};

    const int ln15   = lane & 15;
    const int khalf0 = (lane >> 4) * 16;   // A frag: K base
    const int bk0    = (lane >> 4) * 16;   // B frag: K base

    // Prologue: stage 0 DMA into buffer 0
    if (tid < 32) {
        tdm_load_2d_f16(a_off[0], xrow, IN_F, TOKENS, BK2, BM, IN_F);
        tdm_load_2d_f16(b_off[0], wrow, IN_F, OUT_F, BK2, BN, IN_F);
    }

    const int nstages = IN_F / BK2;   // 64
    for (int s = 0; s < nstages; ++s) {
        const int cur = s & 1;
        const int kt_next = (s + 1) * BK2;

        if (tid < 32) {
            if (kt_next < IN_F) {
                // Prefetch next stage into the other buffer, keep it in flight.
                tdm_load_2d_f16(a_off[cur ^ 1], xrow + kt_next,
                                IN_F, TOKENS, BK2, BM, IN_F);
                tdm_load_2d_f16(b_off[cur ^ 1], wrow + kt_next,
                                IN_F, OUT_F, BK2, BN, IN_F);
                __builtin_amdgcn_s_wait_tensorcnt(2);  // stage s's 2 ops done
            } else {
                __builtin_amdgcn_s_wait_tensorcnt(0);
            }
        }
        __syncthreads();   // all waves see stage s's tiles

        const _Float16* la = &lds_a[cur][0];
        const _Float16* lb = &lds_b[cur][0];

        #pragma unroll
        for (int ks = 0; ks < 2; ++ks) {
            const int kb = ks * 32;
            HV B[4];
            #pragma unroll
            for (int ni = 0; ni < 4; ++ni) {
                const int bn = (waveN * 64 + ni * 16 + ln15) * BK2 + kb;
                B[ni].h[0] = *(const v8h*)&lb[bn + bk0 + 0];
                B[ni].h[1] = *(const v8h*)&lb[bn + bk0 + 8];
            }
            #pragma unroll
            for (int mi = 0; mi < 2; ++mi) {
                const int am = (waveM * 32 + mi * 16 + ln15) * BK2 + kb;
                HV A;
                A.h[0] = *(const v8h*)&la[am + khalf0 + 0];
                A.h[1] = *(const v8h*)&la[am + khalf0 + 8];
                #pragma unroll
                for (int ni = 0; ni < 4; ++ni) {
                    acc[mi][ni] = __builtin_amdgcn_wmma_f32_16x16x32_f16(
                        false, A.v, false, B[ni].v,
                        (short)0, acc[mi][ni], false, false);
                }
            }
        }
        __syncthreads();   // buffer 'cur' free before it is overwritten at s+2
    }

    // Epilogue: per-out-channel scale + bias; non-temporal stores so the
    // 360 MB output stream doesn't evict x'/W' (154 MB resident in 192 MB L2).
    const int rbase = (lane >> 4) * 8;
    #pragma unroll
    for (int ni = 0; ni < 4; ++ni) {
        const int col = n0 + waveN * 64 + ni * 16 + ln15;
        const float s  = scales[col];
        const float bb = bias[col];
        #pragma unroll
        for (int mi = 0; mi < 2; ++mi) {
            const int rowbase = m0 + waveM * 32 + mi * 16 + rbase;
            #pragma unroll
            for (int r = 0; r < 8; ++r) {
                __builtin_nontemporal_store(
                    acc[mi][ni][r] * s + bb,
                    &out[(size_t)(rowbase + r) * OUT_F + col]);
            }
        }
    }
}

// ---------------------------------------------------------------------------
// Fallback: convert-in-kernel version (no workspace needed)
// ---------------------------------------------------------------------------
__global__ __launch_bounds__(256) void qlinear_wmma_direct(
    const float* __restrict__ x,
    const int8_t* __restrict__ w,
    const float* __restrict__ scales,
    const float* __restrict__ bias,
    float* __restrict__ out)
{
    __shared__ _Float16 lds_a[BM * BK];
    __shared__ _Float16 lds_b[BN * BK];

    const int tid   = threadIdx.x;
    const int lane  = tid & 31;
    const int wave  = tid >> 5;
    const int waveM = wave >> 1;
    const int waveN = wave & 1;

    const int m0 = blockIdx.y * BM;
    const int n0 = blockIdx.x * BN;

    const int srow = tid >> 1;
    const int skb  = (tid & 1) * 16;
    const float*  xg = x + (size_t)(m0 + srow) * IN_F + skb;
    const int8_t* wg = w + (size_t)(n0 + srow) * IN_F + skb;

    v8f acc[2][4] = {};

    const int ln15   = lane & 15;
    const int khalf0 = (lane >> 4) * 16;
    const int bk0    = (lane >> 4) * 16;

    for (int kt = 0; kt < IN_F; kt += BK) {
        {
            v4f f0 = *(const v4f*)(xg + kt + 0);
            v4f f1 = *(const v4f*)(xg + kt + 4);
            v4f f2 = *(const v4f*)(xg + kt + 8);
            v4f f3 = *(const v4f*)(xg + kt + 12);
            v8h h0, h1;
            #pragma unroll
            for (int i = 0; i < 4; ++i) {
                h0[i]     = (_Float16)f0[i];
                h0[i + 4] = (_Float16)f1[i];
                h1[i]     = (_Float16)f2[i];
                h1[i + 4] = (_Float16)f3[i];
            }
            *(v8h*)&lds_a[srow * BK + skb + 0] = h0;
            *(v8h*)&lds_a[srow * BK + skb + 8] = h1;
        }
        {
            int4 wv = *(const int4*)(wg + kt);
            const int8_t* wb = (const int8_t*)&wv;
            v8h h0, h1;
            #pragma unroll
            for (int i = 0; i < 8; ++i) {
                h0[i] = (_Float16)(int)wb[i];
                h1[i] = (_Float16)(int)wb[i + 8];
            }
            *(v8h*)&lds_b[srow * BK + skb + 0] = h0;
            *(v8h*)&lds_b[srow * BK + skb + 8] = h1;
        }
        if (kt + BK < IN_F) {
            __builtin_prefetch(xg + kt + BK, 0, 1);
            __builtin_prefetch(wg + kt + BK, 0, 1);
        }

        __syncthreads();

        HV B[4];
        #pragma unroll
        for (int ni = 0; ni < 4; ++ni) {
            const int bn = (waveN * 64 + ni * 16 + ln15) * BK;
            B[ni].h[0] = *(const v8h*)&lds_b[bn + bk0 + 0];
            B[ni].h[1] = *(const v8h*)&lds_b[bn + bk0 + 8];
        }
        #pragma unroll
        for (int mi = 0; mi < 2; ++mi) {
            const int am = (waveM * 32 + mi * 16 + ln15) * BK;
            HV A;
            A.h[0] = *(const v8h*)&lds_a[am + khalf0 + 0];
            A.h[1] = *(const v8h*)&lds_a[am + khalf0 + 8];
            #pragma unroll
            for (int ni = 0; ni < 4; ++ni) {
                acc[mi][ni] = __builtin_amdgcn_wmma_f32_16x16x32_f16(
                    false, A.v, false, B[ni].v,
                    (short)0, acc[mi][ni], false, false);
            }
        }
        __syncthreads();
    }

    const int rbase = (lane >> 4) * 8;
    #pragma unroll
    for (int ni = 0; ni < 4; ++ni) {
        const int col = n0 + waveN * 64 + ni * 16 + ln15;
        const float s  = scales[col];
        const float bb = bias[col];
        #pragma unroll
        for (int mi = 0; mi < 2; ++mi) {
            const int rowbase = m0 + waveM * 32 + mi * 16 + rbase;
            #pragma unroll
            for (int r = 0; r < 8; ++r) {
                out[(size_t)(rowbase + r) * OUT_F + col] = acc[mi][ni][r] * s + bb;
            }
        }
    }
}

extern "C" void kernel_launch(void* const* d_in, const int* in_sizes, int n_in,
                              void* d_out, int out_size, void* d_ws, size_t ws_size,
                              hipStream_t stream) {
    const float*  x      = (const float*)d_in[0];
    const int8_t* wq     = (const int8_t*)d_in[1];
    const float*  scales = (const float*)d_in[2];
    const float*  bias   = (const float*)d_in[3];
    float* out = (float*)d_out;

    dim3 grid(OUT_F / BN, TOKENS / BM);   // 86 x 64 blocks

    const size_t xh_bytes = (size_t)TOKENS * IN_F * sizeof(_Float16);  // 64 MB
    const size_t wh_bytes = (size_t)OUT_F  * IN_F * sizeof(_Float16);  // 90 MB

    if (ws_size >= xh_bytes + wh_bytes) {
        _Float16* xh = (_Float16*)d_ws;
        _Float16* wh = (_Float16*)((char*)d_ws + xh_bytes);

        // one-time converts (stream-ordered ahead of the GEMM)
        cvt_x_f16<<<(TOKENS * (size_t)IN_F) / (256 * 8),  256, 0, stream>>>(x, xh);
        cvt_w_f16<<<(OUT_F  * (size_t)IN_F) / (256 * 16), 256, 0, stream>>>(wq, wh);

        qlinear_wmma_tdm<<<grid, 256, 0, stream>>>(xh, wh, scales, bias, out);
    } else {
        qlinear_wmma_direct<<<grid, 256, 0, stream>>>(x, wq, scales, bias, out);
    }
}
